// HARDPurG_79534204387742
// MI455X (gfx1250) — compile-verified, matrix operand order checked by reading, not run on hardware
//
#include <hip/hip_runtime.h>
#include <stdint.h>

// Problem constants from the reference
#define Bc 16
#define Nc 2048
#define Kt 20
#define NOISE_SCALE 1e-4f

typedef __attribute__((ext_vector_type(2)))  float    v2f;
typedef __attribute__((ext_vector_type(8)))  float    v8f;
typedef __attribute__((ext_vector_type(16))) _Float16 v16h;

// ---------------------------------------------------------------------------
// K1: relu + noise-doped per-row top-20 mask.  One 256-thread block per row.
// Each thread owns 8 contiguous elements (2x float4).  20 rounds of block
// argmax on packed (valbits<<32 | (N-1-idx)) keys; winners marked with -1.
// ---------------------------------------------------------------------------
__global__ void __launch_bounds__(256) k_topk(const float* __restrict__ A,
                                              const float* __restrict__ Nz,
                                              float* __restrict__ Out) {
  const int row = blockIdx.x;           // 0 .. B*N-1 (flat row)
  const int t   = threadIdx.x;          // 0 .. 255
  const size_t base = (size_t)row * Nc;

  const float4* A4 = (const float4*)(A + base);
  const float4* N4 = (const float4*)(Nz + base);
  float4 a0 = A4[2 * t], a1 = A4[2 * t + 1];
  float4 n0 = N4[2 * t], n1 = N4[2 * t + 1];

  float a[8], d[8];
  a[0] = fmaxf(a0.x, 0.f); a[1] = fmaxf(a0.y, 0.f);
  a[2] = fmaxf(a0.z, 0.f); a[3] = fmaxf(a0.w, 0.f);
  a[4] = fmaxf(a1.x, 0.f); a[5] = fmaxf(a1.y, 0.f);
  a[6] = fmaxf(a1.z, 0.f); a[7] = fmaxf(a1.w, 0.f);
  d[0] = a[0] + n0.x * NOISE_SCALE; d[1] = a[1] + n0.y * NOISE_SCALE;
  d[2] = a[2] + n0.z * NOISE_SCALE; d[3] = a[3] + n0.w * NOISE_SCALE;
  d[4] = a[4] + n1.x * NOISE_SCALE; d[5] = a[5] + n1.y * NOISE_SCALE;
  d[6] = a[6] + n1.z * NOISE_SCALE; d[7] = a[7] + n1.w * NOISE_SCALE;

  __shared__ unsigned long long smax[8];
  const int lane = t & 31;
  const int wid  = t >> 5;

  for (int it = 0; it < Kt; ++it) {
    // local argmax over 8 owned elements (doped values are >= 0, so the
    // float bit pattern is monotonic; marked entries (-1) map to key 0)
    unsigned long long key = 0ull;
#pragma unroll
    for (int j = 0; j < 8; ++j) {
      unsigned long long kj =
          (d[j] >= 0.f)
              ? ((((unsigned long long)__float_as_uint(d[j])) << 32) |
                 (unsigned)(Nc - 1 - (t * 8 + j)))
              : 0ull;
      key = (kj > key) ? kj : key;
    }
    // wave32 butterfly reduce
#pragma unroll
    for (int off = 16; off > 0; off >>= 1) {
      unsigned long long o = __shfl_xor(key, off, 32);
      key = (o > key) ? o : key;
    }
    __syncthreads();                 // protect smax reuse from prev round
    if (lane == 0) smax[wid] = key;
    __syncthreads();
    unsigned long long best = smax[0];
#pragma unroll
    for (int w = 1; w < 8; ++w) best = (smax[w] > best) ? smax[w] : best;

    const int widx = (Nc - 1) - (int)(best & 0xffffffffu);
    if ((widx >> 3) == t) {          // owner marks the winner
      const int slot = widx & 7;
#pragma unroll
      for (int j = 0; j < 8; ++j)
        if (j == slot) d[j] = -1.0f;
    }
  }

  float4 o0, o1;
  o0.x = (d[0] < 0.f) ? a[0] : 0.f; o0.y = (d[1] < 0.f) ? a[1] : 0.f;
  o0.z = (d[2] < 0.f) ? a[2] : 0.f; o0.w = (d[3] < 0.f) ? a[3] : 0.f;
  o1.x = (d[4] < 0.f) ? a[4] : 0.f; o1.y = (d[5] < 0.f) ? a[5] : 0.f;
  o1.z = (d[6] < 0.f) ? a[6] : 0.f; o1.w = (d[7] < 0.f) ? a[7] : 0.f;
  ((float4*)(Out + base))[2 * t]     = o0;
  ((float4*)(Out + base))[2 * t + 1] = o1;
}

// ---------------------------------------------------------------------------
// K2: in-place symmetrize (A+A^T)/2 with +I folded into diagonal tiles.
// One block owns the 32x32 tile pair (ti,tj)/(tj,ti); LDS transpose.
// ---------------------------------------------------------------------------
__global__ void __launch_bounds__(256) k_sym(float* __restrict__ M) {
  const int ti = blockIdx.x, tj = blockIdx.y, b = blockIdx.z;
  if (ti > tj) return;

  __shared__ float T0[32][33];
  __shared__ float T1[32][33];

  const int t = threadIdx.x;
  const int r = t >> 3;          // 0..31
  const int c = (t & 7) * 4;     // 0,4,..,28
  const size_t mb = (size_t)b * Nc * Nc;

  const float4 v0 = *(const float4*)(M + mb + (size_t)(ti * 32 + r) * Nc + tj * 32 + c);
  const float4 v1 = *(const float4*)(M + mb + (size_t)(tj * 32 + r) * Nc + ti * 32 + c);
  T0[r][c] = v0.x; T0[r][c + 1] = v0.y; T0[r][c + 2] = v0.z; T0[r][c + 3] = v0.w;
  T1[r][c] = v1.x; T1[r][c + 1] = v1.y; T1[r][c + 2] = v1.z; T1[r][c + 3] = v1.w;
  __syncthreads();

  float4 o0;
  o0.x = 0.5f * (T0[r][c]     + T1[c][r]);
  o0.y = 0.5f * (T0[r][c + 1] + T1[c + 1][r]);
  o0.z = 0.5f * (T0[r][c + 2] + T1[c + 2][r]);
  o0.w = 0.5f * (T0[r][c + 3] + T1[c + 3][r]);
  if (ti == tj) {                 // +I (diagonal lives only in diagonal tiles)
    if (r == c)     o0.x += 1.0f;
    if (r == c + 1) o0.y += 1.0f;
    if (r == c + 2) o0.z += 1.0f;
    if (r == c + 3) o0.w += 1.0f;
  }
  *(float4*)(M + mb + (size_t)(ti * 32 + r) * Nc + tj * 32 + c) = o0;

  if (ti != tj) {
    float4 o1;
    o1.x = 0.5f * (T1[r][c]     + T0[c][r]);
    o1.y = 0.5f * (T1[r][c + 1] + T0[c + 1][r]);
    o1.z = 0.5f * (T1[r][c + 2] + T0[c + 2][r]);
    o1.w = 0.5f * (T1[r][c + 3] + T0[c + 3][r]);
    *(float4*)(M + mb + (size_t)(tj * 32 + r) * Nc + ti * 32 + c) = o1;
  }
}

// ---------------------------------------------------------------------------
// K3: row sums via WMMA (tile x all-ones), then d^{-1/2} into d_ws.
// Each wave handles 16 rows; C accumulates across all 128 K-tiles, so the
// inner loop is a pure back-to-back v_wmma_f32_16x16x4_f32 chain.
// A-operand layout (ISA 7.12.2): lane m (0-15) holds row m, K=0/1 in its
// 2 VGPRs; lane m+16 holds K=2/3.  With B == ones the column assignment is
// free, so lane l simply streams contiguous float4 pairs from row (l&15).
// ---------------------------------------------------------------------------
__global__ void __launch_bounds__(128) k_rowsum(const float* __restrict__ M,
                                                float* __restrict__ dinv) {
  const int t    = threadIdx.x;
  const int lane = t & 31;
  const int w    = t >> 5;
  const int rowBase = blockIdx.x * 64 + w * 16;   // 16 rows per wave
  const int r    = lane & 15;
  const int half = lane >> 4;
  const float* rowPtr = M + (size_t)(rowBase + r) * Nc;

  v8f acc = {0.f, 0.f, 0.f, 0.f, 0.f, 0.f, 0.f, 0.f};

#if __has_builtin(__builtin_amdgcn_wmma_f32_16x16x4_f32)
  const v2f ones = {1.0f, 1.0f};
  for (int tile = 0; tile < Nc / 16; ++tile) {
    const float4* p = (const float4*)(rowPtr + tile * 16 + 8 * half);
    const float4 q0 = p[0], q1 = p[1];
    v2f a0 = {q0.x, q0.y}, a1 = {q0.z, q0.w};
    v2f a2 = {q1.x, q1.y}, a3 = {q1.z, q1.w};
    acc = __builtin_amdgcn_wmma_f32_16x16x4_f32(false, a0, false, ones, (short)0, acc, false, false);
    acc = __builtin_amdgcn_wmma_f32_16x16x4_f32(false, a1, false, ones, (short)0, acc, false, false);
    acc = __builtin_amdgcn_wmma_f32_16x16x4_f32(false, a2, false, ones, (short)0, acc, false, false);
    acc = __builtin_amdgcn_wmma_f32_16x16x4_f32(false, a3, false, ones, (short)0, acc, false, false);
  }
#else
  // Fallback: probe-confirmed f16 WMMA (ones trick, 32-column tiles).
  v16h ones16;
#pragma unroll
  for (int i = 0; i < 16; ++i) ones16[i] = (_Float16)1.0f;
  for (int tile = 0; tile < Nc / 32; ++tile) {
    const float4* p = (const float4*)(rowPtr + tile * 32 + 16 * half);
    const float4 q0 = p[0], q1 = p[1], q2 = p[2], q3 = p[3];
    float q[16] = {q0.x, q0.y, q0.z, q0.w, q1.x, q1.y, q1.z, q1.w,
                   q2.x, q2.y, q2.z, q2.w, q3.x, q3.y, q3.z, q3.w};
    v16h h;
#pragma unroll
    for (int i = 0; i < 16; ++i) h[i] = (_Float16)q[i];
    acc = __builtin_amdgcn_wmma_f32_16x16x32_f16(false, h, false, ones16, (short)0, acc, false, false);
  }
#endif

  // D[m, n] == rowsum(m) for all n.  Lane 0 holds rows 0-7 in acc[0..7],
  // lane 16 holds rows 8-15 (C/D layout, ISA 7.12.2).
  if ((lane & 15) == 0) {
#pragma unroll
    for (int v = 0; v < 8; ++v) {
      const float s = acc[v];
      dinv[rowBase + 8 * half + v] = (s > 0.f) ? rsqrtf(s) : 0.f;
    }
  }
}

// ---------------------------------------------------------------------------
// K4: A[b,i,j] *= dinv[b,i] * dinv[b,j]   (float4 streaming)
// ---------------------------------------------------------------------------
__global__ void __launch_bounds__(256) k_scale(float* __restrict__ M,
                                               const float* __restrict__ dinv) {
  const size_t idx = (size_t)blockIdx.x * 256 + threadIdx.x;  // float4 index
  const size_t e   = idx * 4;
  const size_t R   = e >> 11;               // flat row b*N+i
  const int    j   = (int)(e & (Nc - 1));
  const int    b   = (int)(R >> 11);

  const float  di = dinv[R];
  const float4 dj = *(const float4*)(dinv + ((size_t)b << 11) + j);
  float4 v = *(float4*)(M + e);
  v.x *= di * dj.x; v.y *= di * dj.y; v.z *= di * dj.z; v.w *= di * dj.w;
  *(float4*)(M + e) = v;
}

// ---------------------------------------------------------------------------
extern "C" void kernel_launch(void* const* d_in, const int* in_sizes, int n_in,
                              void* d_out, int out_size, void* d_ws, size_t ws_size,
                              hipStream_t stream) {
  const float* A  = (const float*)d_in[0];   // (16, 2048, 2048) f32
  const float* Nz = (const float*)d_in[1];   // (16, 2048, 2048) f32
  float* Out  = (float*)d_out;               // (16, 2048, 2048) f32, also working buf
  float* dinv = (float*)d_ws;                // B*N floats = 128 KB scratch

  // 1) relu + top-20 sparsify  -> Out
  k_topk<<<dim3(Bc * Nc), dim3(256), 0, stream>>>(A, Nz, Out);
  // 2) symmetrize in-place (+I on diagonal tiles)
  k_sym<<<dim3(Nc / 32, Nc / 32, Bc), dim3(256), 0, stream>>>(Out);
  // 3) row sums via WMMA -> d^{-1/2}
  k_rowsum<<<dim3((Bc * Nc) / 64), dim3(128), 0, stream>>>(Out, dinv);
  // 4) D^{-1/2} A D^{-1/2}
  k_scale<<<dim3((unsigned)((size_t)Bc * Nc * Nc / 4 / 256)), dim3(256), 0, stream>>>(Out, dinv);
}
